// GCN_94489280637
// MI455X (gfx1250) — matrix operand order repaired and checked by reading, not compiled, not run on hardware
//
#include <hip/hip_runtime.h>
#include <hip/hip_bf16.h>
#include <math.h>

// ---------------------------------------------------------------------------
// GCN forward for MI455X (gfx1250, wave32, WMMA).
//   h      = relu( spmm(x @ W1) + b1 )
//   logits = spmm(h @ W2) + b2
//   out    = log_softmax(logits)
//
// v2 changes vs v1 (reasoned from MI455X specs, compile-only):
//  * SpMM via dst-CSR (counting sort: hist + 2-level scan + scatter) and
//    register gather-reduce -> removes ~950M f32 global atomics (the L2
//    atomic-throughput ceiling), replaces them with plain L2-resident reads.
//  * GEMM grids reordered so all N-slab blocks of the same rows launch
//    adjacently -> A matrix streams from HBM once, slab re-reads hit L2.
//  * W staged in LDS pre-split into bf16 hi/lo planes (padded transposed
//    layout) -> B operand = 2x ds_load_b128 + shuffle per k-step.
//  * Still 3x V_WMMA_F32_16X16X32_BF16 per k-step (hi/lo split ~ fp32 acc).
// ---------------------------------------------------------------------------

typedef __attribute__((ext_vector_type(16))) __bf16 v16bf;
typedef __attribute__((ext_vector_type(8)))  __bf16 v8bf;
typedef __attribute__((ext_vector_type(8)))  float  v8f;

#define NFEAT  512
#define NHID   256
#define NCLASS 40
#define PAD1   (NFEAT + 8)   // LDS row pitch (halfwords) for gemm1 W slab
#define PAD2   (NHID + 8)    // LDS row pitch for gemm2 W slab

__device__ __forceinline__ v16bf cat16(v8bf a, v8bf b) {
    return __builtin_shufflevector(a, b, 0,1,2,3,4,5,6,7,8,9,10,11,12,13,14,15);
}

// ---------------------------------------------------------------- utilities
__global__ __launch_bounds__(256) void zero_int_kernel(int* __restrict__ p, int n) {
    int i = blockIdx.x * blockDim.x + threadIdx.x;
    if (i < n) p[i] = 0;
}

// ------------------------------------------------- CSR build: degree histogram
__global__ __launch_bounds__(256) void hist_kernel(const int* __restrict__ edst,
                                                   int* __restrict__ deg, int E) {
    int e = blockIdx.x * blockDim.x + threadIdx.x;
    if (e < E) atomicAdd(&deg[edst[e]], 1);
}

// -------------------------------------------- scan level 1: per-chunk totals
__global__ __launch_bounds__(256) void scan_reduce_kernel(const int* __restrict__ deg,
                                                          int* __restrict__ csum, int Nn) {
    __shared__ int s[256];
    const int b = blockIdx.x, t = threadIdx.x;
    const int base = b * 1024;
    int acc = 0;
    for (int i = t; i < 1024; i += 256) {
        int idx = base + i;
        acc += (idx < Nn) ? deg[idx] : 0;
    }
    s[t] = acc; __syncthreads();
    for (int o = 128; o > 0; o >>= 1) {
        if (t < o) s[t] += s[t + o];
        __syncthreads();
    }
    if (t == 0) csum[b] = s[0];
}

// ------------------------------------- scan level 2: serial scan of chunk sums
__global__ void scan_top_kernel(int* __restrict__ csum, int B) {
    if (blockIdx.x == 0 && threadIdx.x == 0) {
        int run = 0;
        for (int i = 0; i < B; ++i) { int t = csum[i]; csum[i] = run; run += t; }
    }
}

// ------------------- scan level 3: per-chunk exclusive scan -> off[] and cur[]
__global__ __launch_bounds__(256) void scan_block_kernel(const int* __restrict__ deg,
                                                         const int* __restrict__ csum,
                                                         int* __restrict__ off,
                                                         int* __restrict__ cur, int Nn) {
    __shared__ int ts[256];
    const int b = blockIdx.x, t = threadIdx.x;
    const int base = b * 1024 + t * 4;
    int v0 = (base + 0 < Nn) ? deg[base + 0] : 0;
    int v1 = (base + 1 < Nn) ? deg[base + 1] : 0;
    int v2 = (base + 2 < Nn) ? deg[base + 2] : 0;
    int v3 = (base + 3 < Nn) ? deg[base + 3] : 0;
    int tot = v0 + v1 + v2 + v3;
    ts[t] = tot; __syncthreads();
    for (int o = 1; o < 256; o <<= 1) {        // Hillis-Steele inclusive scan
        int x = (t >= o) ? ts[t - o] : 0;
        __syncthreads();
        ts[t] += x;
        __syncthreads();
    }
    int excl = ts[t] - tot + csum[b];
    int o0 = excl, o1 = o0 + v0, o2 = o1 + v1, o3 = o2 + v2;
    if (base + 0 < Nn) { off[base + 0] = o0; cur[base + 0] = o0; }
    if (base + 1 < Nn) { off[base + 1] = o1; cur[base + 1] = o1; }
    if (base + 2 < Nn) { off[base + 2] = o2; cur[base + 2] = o2; }
    if (base + 3 < Nn) { off[base + 3] = o3; cur[base + 3] = o3; }
}

// -------------------------------------------- scatter edges into dst-CSR slots
__global__ __launch_bounds__(256) void scatter_kernel(const int* __restrict__ esrc,
                                                      const int* __restrict__ edst,
                                                      const float* __restrict__ eval,
                                                      int* __restrict__ cur,
                                                      int* __restrict__ csr_src,
                                                      float* __restrict__ csr_val, int E) {
    int e = blockIdx.x * blockDim.x + threadIdx.x;
    if (e >= E) return;
    int pos = atomicAdd(&cur[edst[e]], 1);
    csr_src[pos] = esrc[e];
    csr_val[pos] = eval[e];
}

// ------------------------------------------------------- GEMM1: H1 = X @ W1
// grid = (16 N-slabs, M-tile groups): slab blocks of the same rows launch
// adjacently so X streams from HBM once and slab re-reads hit L2.
__global__ __launch_bounds__(256) void gemm1_kernel(const float* __restrict__ X,
                                                    const float* __restrict__ W1,
                                                    float* __restrict__ H1,
                                                    int MT) {
    __shared__ __bf16 sWhi[16 * PAD1];
    __shared__ __bf16 sWlo[16 * PAD1];
    const int n0 = blockIdx.x * 16;
    for (int i = threadIdx.x; i < NFEAT * 16; i += 256) {
        int k = i >> 4, n = i & 15;
        float w = W1[k * NHID + n0 + n];
        __bf16 h = (__bf16)w;
        sWhi[n * PAD1 + k] = h;
        sWlo[n * PAD1 + k] = (__bf16)(w - (float)h);
    }
    __syncthreads();

    const int lane = threadIdx.x & 31;
    const int half = lane >> 4;     // K-half select
    const int l15  = lane & 15;     // M (for A) / N (for B,C)
    const int mt   = blockIdx.y * 8 + (threadIdx.x >> 5);
    if (mt >= MT) return;

    const float* arow = X + (size_t)(mt * 16 + l15) * NFEAT;
    v8f acc = {};

    for (int k0 = 0; k0 < NFEAT; k0 += 32) {
        // ---- A tile 16x32: lane half 0 -> K {0..7,16..23}, half 1 -> +8 ----
        float av[16];
        {
            const float4* p0 = (const float4*)(arow + k0 + half * 8);
            const float4* p1 = (const float4*)(arow + k0 + half * 8 + 16);
            float4 f0 = p0[0], f1 = p0[1], g0 = p1[0], g1 = p1[1];
            av[0]=f0.x;  av[1]=f0.y;  av[2]=f0.z;  av[3]=f0.w;
            av[4]=f1.x;  av[5]=f1.y;  av[6]=f1.z;  av[7]=f1.w;
            av[8]=g0.x;  av[9]=g0.y;  av[10]=g0.z; av[11]=g0.w;
            av[12]=g1.x; av[13]=g1.y; av[14]=g1.z; av[15]=g1.w;
        }
        v16bf ahi, alo;
#pragma unroll
        for (int i = 0; i < 16; ++i) {
            __bf16 h = (__bf16)av[i];
            ahi[i] = h;
            alo[i] = (__bf16)(av[i] - (float)h);
        }
        // ---- B tile 32x16: lane half 0 -> K {0..7,16..23}, half 1 -> +8 ----
        const int kb = k0 + half * 8;
        const __bf16* ph = sWhi + l15 * PAD1 + kb;
        const __bf16* pl = sWlo + l15 * PAD1 + kb;
        v16bf bhi = cat16(*(const v8bf*)ph, *(const v8bf*)(ph + 16));
        v16bf blo = cat16(*(const v8bf*)pl, *(const v8bf*)(pl + 16));

        acc = __builtin_amdgcn_wmma_f32_16x16x32_bf16(false, ahi, false, bhi, (short)0, acc, false, false);
        acc = __builtin_amdgcn_wmma_f32_16x16x32_bf16(false, alo, false, bhi, (short)0, acc, false, false);
        acc = __builtin_amdgcn_wmma_f32_16x16x32_bf16(false, ahi, false, blo, (short)0, acc, false, false);
    }

    const int col = n0 + l15;
#pragma unroll
    for (int r = 0; r < 8; ++r) {
        int row = mt * 16 + r + half * 8;
        H1[(size_t)row * NHID + col] = acc[r];
    }
}

// ------------------------- SpMM1 (gather): AH1[n] = sum_e val[e] * H1[src[e]]
// One wave per dst node; lane holds 8 contiguous features in registers.
// No float atomics; H1 (102MB) is L2-resident on the 192MB L2.
__global__ __launch_bounds__(256) void spmm_gather1_kernel(const int* __restrict__ off,
                                                           const int* __restrict__ deg,
                                                           const int* __restrict__ csr_src,
                                                           const float* __restrict__ csr_val,
                                                           const float* __restrict__ H,
                                                           float* __restrict__ out, int Nn) {
    int wid  = (int)(((long long)blockIdx.x * blockDim.x + threadIdx.x) >> 5);
    int lane = threadIdx.x & 31;
    if (wid >= Nn) return;
    const int beg = off[wid], cnt = deg[wid];
    float a0=0.f,a1=0.f,a2=0.f,a3=0.f,a4=0.f,a5=0.f,a6=0.f,a7=0.f;
    for (int j = 0; j < cnt; ++j) {
        int   s = csr_src[beg + j];
        float v = csr_val[beg + j];
        const float4* hp = (const float4*)(H + (size_t)s * NHID) + lane * 2;
        float4 h0 = hp[0], h1 = hp[1];
        a0 = fmaf(v, h0.x, a0); a1 = fmaf(v, h0.y, a1);
        a2 = fmaf(v, h0.z, a2); a3 = fmaf(v, h0.w, a3);
        a4 = fmaf(v, h1.x, a4); a5 = fmaf(v, h1.y, a5);
        a6 = fmaf(v, h1.z, a6); a7 = fmaf(v, h1.w, a7);
    }
    float4* op = (float4*)(out + (size_t)wid * NHID) + lane * 2;
    float4 o0 = {a0, a1, a2, a3}, o1 = {a4, a5, a6, a7};
    op[0] = o0; op[1] = o1;
}

// ---------------------------- GEMM2: H2 = relu(AH1 + b1) @ W2   [M,256]x[256,40]
__global__ __launch_bounds__(256) void gemm2_kernel(const float* __restrict__ AH1,
                                                    const float* __restrict__ b1,
                                                    const float* __restrict__ W2,
                                                    float* __restrict__ H2,
                                                    int MT) {
    __shared__ __bf16 sWhi[16 * PAD2];
    __shared__ __bf16 sWlo[16 * PAD2];
    __shared__ float  sB[NHID];
    const int n0 = blockIdx.x * 16;
    for (int i = threadIdx.x; i < NHID * 16; i += 256) {
        int k = i >> 4, n = i & 15;
        float w = (n0 + n < NCLASS) ? W2[k * NCLASS + n0 + n] : 0.0f;
        __bf16 h = (__bf16)w;
        sWhi[n * PAD2 + k] = h;
        sWlo[n * PAD2 + k] = (__bf16)(w - (float)h);
    }
    for (int i = threadIdx.x; i < NHID; i += 256) sB[i] = b1[i];
    __syncthreads();

    const int lane = threadIdx.x & 31;
    const int half = lane >> 4;
    const int l15  = lane & 15;
    const int mt   = blockIdx.y * 8 + (threadIdx.x >> 5);
    if (mt >= MT) return;

    const float* arow = AH1 + (size_t)(mt * 16 + l15) * NHID;
    v8f acc = {};

    for (int k0 = 0; k0 < NHID; k0 += 32) {
        const int ka = k0 + half * 8;
        float av[16];
        {
            const float4* p0 = (const float4*)(arow + ka);
            const float4* p1 = (const float4*)(arow + ka + 16);
            float4 f0 = p0[0], f1 = p0[1], g0 = p1[0], g1 = p1[1];
            av[0]=f0.x;  av[1]=f0.y;  av[2]=f0.z;  av[3]=f0.w;
            av[4]=f1.x;  av[5]=f1.y;  av[6]=f1.z;  av[7]=f1.w;
            av[8]=g0.x;  av[9]=g0.y;  av[10]=g0.z; av[11]=g0.w;
            av[12]=g1.x; av[13]=g1.y; av[14]=g1.z; av[15]=g1.w;
        }
        v16bf ahi, alo;
#pragma unroll
        for (int i = 0; i < 16; ++i) {
            int k = (i < 8) ? (ka + i) : (ka + 8 + i);   // +b1, ReLU (fused)
            float a = fmaxf(av[i] + sB[k], 0.0f);
            __bf16 h = (__bf16)a;
            ahi[i] = h;
            alo[i] = (__bf16)(a - (float)h);
        }
        const int kb = k0 + half * 8;
        const __bf16* ph = sWhi + l15 * PAD2 + kb;
        const __bf16* pl = sWlo + l15 * PAD2 + kb;
        v16bf bhi = cat16(*(const v8bf*)ph, *(const v8bf*)(ph + 16));
        v16bf blo = cat16(*(const v8bf*)pl, *(const v8bf*)(pl + 16));

        acc = __builtin_amdgcn_wmma_f32_16x16x32_bf16(false, ahi, false, bhi, (short)0, acc, false, false);
        acc = __builtin_amdgcn_wmma_f32_16x16x32_bf16(false, alo, false, bhi, (short)0, acc, false, false);
        acc = __builtin_amdgcn_wmma_f32_16x16x32_bf16(false, ahi, false, blo, (short)0, acc, false, false);
    }

    const int col = n0 + l15;
    if (col < NCLASS) {
#pragma unroll
        for (int r = 0; r < 8; ++r) {
            int row = mt * 16 + r + half * 8;
            H2[(size_t)row * NCLASS + col] = acc[r];
        }
    }
}

// --------------------- SpMM2 (gather): LG[n] = sum_e val[e] * H2[src[e]] (40)
__global__ __launch_bounds__(256) void spmm_gather2_kernel(const int* __restrict__ off,
                                                           const int* __restrict__ deg,
                                                           const int* __restrict__ csr_src,
                                                           const float* __restrict__ csr_val,
                                                           const float* __restrict__ H2,
                                                           float* __restrict__ out, int Nn) {
    int wid  = (int)(((long long)blockIdx.x * blockDim.x + threadIdx.x) >> 5);
    int lane = threadIdx.x & 31;
    if (wid >= Nn) return;
    const int beg = off[wid], cnt = deg[wid];
    float a0 = 0.f, a1 = 0.f;
    for (int j = 0; j < cnt; ++j) {
        int   s = csr_src[beg + j];
        float v = csr_val[beg + j];
        const float* hr = H2 + (size_t)s * NCLASS;
        a0 = fmaf(v, hr[lane], a0);
        if (lane < NCLASS - 32) a1 = fmaf(v, hr[32 + lane], a1);
    }
    out[(size_t)wid * NCLASS + lane] = a0;
    if (lane < NCLASS - 32) out[(size_t)wid * NCLASS + 32 + lane] = a1;
}

// --------------------------------- out = log_softmax(LG + b2), one thread/node
__global__ __launch_bounds__(256) void logsoftmax_kernel(const float* __restrict__ LG,
                                                         const float* __restrict__ b2,
                                                         float* __restrict__ out,
                                                         int Nn) {
    int n = blockIdx.x * blockDim.x + threadIdx.x;
    if (n >= Nn) return;
    float v[NCLASS];
    float mx = -3.402823466e+38f;
#pragma unroll
    for (int f = 0; f < NCLASS; ++f) {
        v[f] = LG[(size_t)n * NCLASS + f] + b2[f];
        mx = fmaxf(mx, v[f]);
    }
    float s = 0.0f;
#pragma unroll
    for (int f = 0; f < NCLASS; ++f) s += expf(v[f] - mx);
    float ls = logf(s) + mx;
#pragma unroll
    for (int f = 0; f < NCLASS; ++f) out[(size_t)n * NCLASS + f] = v[f] - ls;
}

// ---------------------------------------------------------------------------
extern "C" void kernel_launch(void* const* d_in, const int* in_sizes, int n_in,
                              void* d_out, int out_size, void* d_ws, size_t ws_size,
                              hipStream_t stream) {
    const float* X    = (const float*)d_in[0];
    const int*   esrc = (const int*)  d_in[1];
    const int*   edst = (const int*)  d_in[2];
    const float* eval = (const float*)d_in[3];
    const float* W1   = (const float*)d_in[4];
    const float* b1   = (const float*)d_in[5];
    const float* W2   = (const float*)d_in[6];
    const float* b2   = (const float*)d_in[7];

    const int Nn = in_sizes[0] / NFEAT;   // 100000
    const int E  = in_sizes[1];           // 3200000

    // ---- workspace partition: H1 | AH1 | H2 | LG | csr_val | csr_src | deg | off | cur | csum
    float* H1      = (float*)d_ws;
    float* AH1     = H1      + (size_t)Nn * NHID;
    float* H2      = AH1     + (size_t)Nn * NHID;
    float* LG      = H2      + (size_t)Nn * NCLASS;
    float* csr_val = LG      + (size_t)Nn * NCLASS;
    int*   csr_src = (int*)(csr_val + (size_t)E);
    int*   deg     = csr_src + (size_t)E;
    int*   off     = deg     + (size_t)Nn;
    int*   cur     = off     + (size_t)Nn;
    int*   csum    = cur     + (size_t)Nn;

    const int NB = (Nn + 1023) / 1024;          // scan chunks (98)
    const unsigned gridE = (unsigned)((E + 255) / 256);
    const unsigned gridW = (unsigned)(((long long)Nn * 32 + 255) / 256);  // wave/node

    // ---- build dst-CSR (counting sort; int atomics only) ----
    zero_int_kernel<<<(unsigned)((Nn + 255) / 256), 256, 0, stream>>>(deg, Nn);
    hist_kernel<<<gridE, 256, 0, stream>>>(edst, deg, E);
    scan_reduce_kernel<<<(unsigned)NB, 256, 0, stream>>>(deg, csum, Nn);
    scan_top_kernel<<<1, 64, 0, stream>>>(csum, NB);
    scan_block_kernel<<<(unsigned)NB, 256, 0, stream>>>(deg, csum, off, cur, Nn);
    scatter_kernel<<<gridE, 256, 0, stream>>>(esrc, edst, eval, cur, csr_src, csr_val, E);

    const int MT = (Nn + 15) / 16;              // 6250 M tiles (exact)

    // ---- layer 1 ----
    dim3 g1(NHID / 16, (MT + 7) / 8);           // (16 slabs, 782): slabs adjacent -> X streams once
    gemm1_kernel<<<g1, 256, 0, stream>>>(X, W1, H1, MT);
    spmm_gather1_kernel<<<gridW, 256, 0, stream>>>(off, deg, csr_src, csr_val, H1, AH1, Nn);

    // ---- layer 2 ----
    dim3 g2((NCLASS + 15) / 16, (MT + 7) / 8);  // (3 slabs, 782)
    gemm2_kernel<<<g2, 256, 0, stream>>>(AH1, b1, W2, H2, MT);
    spmm_gather2_kernel<<<gridW, 256, 0, stream>>>(off, deg, csr_src, csr_val, H2, LG, Nn);

    logsoftmax_kernel<<<(unsigned)((Nn + 255) / 256), 256, 0, stream>>>(LG, b2, (float*)d_out, Nn);
}
